// PEERLayer_42614665511416
// MI455X (gfx1250) — matrix operand order repaired and checked by reading, not compiled, not run on hardware
//
#include <hip/hip_runtime.h>
#include <hip/hip_bf16.h>
#include <math.h>

// ---------------------------------------------------------------------------
// PEER layer for MI455X (gfx1250, wave32).
//  k1: q = x @ Wq           (WMMA bf16 hi/lo split, f32 accum)
//  k2: fused LN + product-key routing + 16-expert FFN (one wave / token-head)
//      centroid keys staged via async global->LDS DMA (ASYNCcnt path)
//  k3: out = y @ Wout       (same WMMA GEMM)
// Workspace: q (16MB f32) + y (16MB f32) = 32MB.
// ---------------------------------------------------------------------------

#define Bb 2
#define Ss 2048
#define Dd 1024
#define Hh 16
#define HD 64
#define SQRTN 256
#define SUBD 32
#define KTOP 16
#define MROWS (Bb * Ss)      // 4096
#define NTH   (MROWS * Hh)   // 65536 token-heads

typedef __attribute__((ext_vector_type(16))) __bf16 v16bf;
typedef __attribute__((ext_vector_type(8)))  float  v8f;

union Frag { unsigned int i[8]; v16bf v; };

__device__ __forceinline__ unsigned short f32_to_bf16(float f) {
  unsigned u = __builtin_bit_cast(unsigned, f);
  unsigned r = u + 0x7FFFu + ((u >> 16) & 1u);   // round-to-nearest-even
  return (unsigned short)(r >> 16);
}
__device__ __forceinline__ float bf16f(unsigned short h) {
  unsigned u = ((unsigned)h) << 16;
  return __builtin_bit_cast(float, u);
}

// Async 16B global->LDS copy (per active lane), tracked by ASYNCcnt.
__device__ __forceinline__ void async_copy_b128(unsigned lds_off,
                                                unsigned long long gaddr) {
  asm volatile("global_load_async_to_lds_b128 %0, %1, off"
               :: "v"(lds_off), "v"(gaddr) : "memory");
}
__device__ __forceinline__ void wait_asynccnt0() {
  asm volatile("s_wait_asynccnt 0x0" ::: "memory");
}

// ---------------------------------------------------------------------------
// WMMA GEMM: C[M,N] = A[M,K] * B[K,N], all f32 in memory.
// Block = 256 threads (8 waves), tile 128(M) x 128(N), K-step 32.
// f32 -> bf16 hi/lo split on the fly during LDS staging; 3 WMMAs per tile.
// LDS ushort layout, row pitch 34 ushorts (17 dwords, odd -> no bank conflict).
// ---------------------------------------------------------------------------
__global__ __launch_bounds__(256)
void peer_gemm_bf16x3(const float* __restrict__ A, const float* __restrict__ Bm,
                      float* __restrict__ C, int M, int N, int K) {
  __shared__ unsigned short Ah[128 * 34];
  __shared__ unsigned short Al[128 * 34];
  __shared__ unsigned short Bh[128 * 34];
  __shared__ unsigned short Bl[128 * 34];

  const int t    = threadIdx.x;
  const int lane = t & 31;
  const int wv   = t >> 5;
  const int m0   = blockIdx.y * 128;
  const int n0   = blockIdx.x * 128;

  const int rowA = t >> 1;            // 0..127
  const int khA  = (t & 1) * 16;      // 0 or 16
  const int kkB  = t & 31;            // 0..31
  const int nbB  = (t >> 5) * 16;     // 0..112

  v8f acc[8];
#pragma unroll
  for (int j = 0; j < 8; ++j) acc[j] = (v8f){0.f,0.f,0.f,0.f,0.f,0.f,0.f,0.f};

  const int mloc = lane & 15;         // row/col within 16
  const int ksel = lane >> 4;         // K-half select

  for (int k0 = 0; k0 < K; k0 += 32) {
    __syncthreads();
    // ---- stage A tile 128x32 (each thread: one row-half, 16 floats) ----
    {
      const float4* ag = (const float4*)(A + (size_t)(m0 + rowA) * K + k0 + khA);
      unsigned short* ph = &Ah[rowA * 34 + khA];
      unsigned short* pl = &Al[rowA * 34 + khA];
#pragma unroll
      for (int i4 = 0; i4 < 4; ++i4) {
        float4 f = ag[i4];
        float vv[4] = {f.x, f.y, f.z, f.w};
#pragma unroll
        for (int c = 0; c < 4; ++c) {
          unsigned short h = f32_to_bf16(vv[c]);
          ph[i4 * 4 + c] = h;
          pl[i4 * 4 + c] = f32_to_bf16(vv[c] - bf16f(h));
        }
      }
    }
    // ---- stage B tile 32x128 transposed (thread: one k-row, 16 cols) ----
    {
      const float4* bg = (const float4*)(Bm + (size_t)(k0 + kkB) * N + n0 + nbB);
#pragma unroll
      for (int i4 = 0; i4 < 4; ++i4) {
        float4 f = bg[i4];
        float vv[4] = {f.x, f.y, f.z, f.w};
#pragma unroll
        for (int c = 0; c < 4; ++c) {
          unsigned short h = f32_to_bf16(vv[c]);
          Bh[(nbB + i4 * 4 + c) * 34 + kkB] = h;
          Bl[(nbB + i4 * 4 + c) * 34 + kkB] = f32_to_bf16(vv[c] - bf16f(h));
        }
      }
    }
    __syncthreads();

    // ---- A fragment for this wave (rows wv*16 .. +15), ISA 16x32 layout ----
    const unsigned* AhW = (const unsigned*)Ah;
    const unsigned* AlW = (const unsigned*)Al;
    const unsigned* BhW = (const unsigned*)Bh;
    const unsigned* BlW = (const unsigned*)Bl;

    Frag fah, fal;
    {
      const int base = (wv * 16 + mloc) * 17 + ksel * 4;
#pragma unroll
      for (int d = 0; d < 8; ++d) {
        const int idx = base + (d >> 2) * 8 + (d & 3);
        fah.i[d] = AhW[idx];
        fal.i[d] = AlW[idx];
      }
    }
#pragma unroll
    for (int j = 0; j < 8; ++j) {
      Frag fbh, fbl;
      const int base = (j * 16 + mloc) * 17 + ksel * 4;
#pragma unroll
      for (int d = 0; d < 8; ++d) {
        fbh.i[d] = BhW[base + (d >> 2) * 8 + (d & 3)];
        fbl.i[d] = BlW[base + (d >> 2) * 8 + (d & 3)];
      }
      acc[j] = __builtin_amdgcn_wmma_f32_16x16x32_bf16(false, fah.v, false, fbh.v,
                                                       (short)0, acc[j], false, false);
      acc[j] = __builtin_amdgcn_wmma_f32_16x16x32_bf16(false, fah.v, false, fbl.v,
                                                       (short)0, acc[j], false, false);
      acc[j] = __builtin_amdgcn_wmma_f32_16x16x32_bf16(false, fal.v, false, fbh.v,
                                                       (short)0, acc[j], false, false);
    }
  }

  // ---- write C: lane holds n = lane%16, m = 8*ksel + r (ISA C/D layout) ----
  const int mBase = m0 + wv * 16 + 8 * ksel;
#pragma unroll
  for (int j = 0; j < 8; ++j) {
    const int nIdx = n0 + j * 16 + mloc;
#pragma unroll
    for (int r = 0; r < 8; ++r)
      C[(size_t)(mBase + r) * N + nIdx] = acc[j][r];
  }
}

// ---------------------------------------------------------------------------
// Wave-cooperative top-16 of 256 values (8 per lane), values+global indices.
// Global index of sc[tt] is (lane + 32*tt). Lexicographic tie-break on index.
// ---------------------------------------------------------------------------
__device__ __forceinline__ void top16_256(float sc[8], int lane,
                                          float tv[16], int ti[16]) {
#pragma unroll 1
  for (int r = 0; r < 16; ++r) {
    float bv = sc[0]; int bt = 0;
#pragma unroll
    for (int tt = 1; tt < 8; ++tt)
      if (sc[tt] > bv) { bv = sc[tt]; bt = tt; }
    int bi = lane + 32 * bt;
#pragma unroll
    for (int o = 16; o >= 1; o >>= 1) {
      float ov = __shfl_xor(bv, o, 32);
      int   oi = __shfl_xor(bi, o, 32);
      if (ov > bv || (ov == bv && oi < bi)) { bv = ov; bi = oi; }
    }
    tv[r] = bv; ti[r] = bi;
    if ((bi & 31) == lane) sc[bi >> 5] = -3.0e38f;
  }
}

// ---------------------------------------------------------------------------
// Fused LN + routing + expert FFN. One wave per token-head; 8 per block.
// Centroid keys staged into one reused 32KB LDS buffer via async global->LDS
// DMA (overlapped with the per-wave LayerNorm), two phases.
// ---------------------------------------------------------------------------
__global__ __launch_bounds__(256)
void peer_router_ffn(const float* __restrict__ x, const float* __restrict__ q,
                     const float* __restrict__ ln_g, const float* __restrict__ ln_b,
                     const float* __restrict__ ckeys, const float* __restrict__ cpkeys,
                     const float* __restrict__ w_down, const float* __restrict__ w_up,
                     float* __restrict__ y) {
  __shared__ float keybuf[SQRTN * SUBD];   // 32 KB, reused for both key tables
  __shared__ float qnb[8][64];
  __shared__ float xhb[8][64];

  const int t = threadIdx.x;
  const int lane = t & 31;
  const int wv = t >> 5;
  const int th = blockIdx.x * 8 + wv;
  const int bs = th >> 4;
  const int h  = th & 15;

  const unsigned keybase = (unsigned)(size_t)(&keybuf[0]);

  // phase A: async-stage c_keys (8 x b128 per thread) while waves do LayerNorm
#pragma unroll
  for (int i = t; i < SQRTN * SUBD / 4; i += 256)
    async_copy_b128(keybase + (unsigned)i * 16u,
                    (unsigned long long)(const void*)ckeys + (unsigned long long)i * 16u);

  {
    const float* qrow = q + (size_t)bs * Dd + h * HD;
    float qa = qrow[lane], qb = qrow[lane + 32];
    float s = qa + qb;
#pragma unroll
    for (int o = 16; o >= 1; o >>= 1) s += __shfl_xor(s, o, 32);
    const float mu = s * (1.f / 64.f);
    const float da = qa - mu, db = qb - mu;
    float v2 = da * da + db * db;
#pragma unroll
    for (int o = 16; o >= 1; o >>= 1) v2 += __shfl_xor(v2, o, 32);
    const float rstd = rsqrtf(v2 * (1.f / 64.f) + 1e-5f);
    qnb[wv][lane]      = da * rstd * ln_g[lane]      + ln_b[lane];
    qnb[wv][lane + 32] = db * rstd * ln_g[lane + 32] + ln_b[lane + 32];
    const float* xrow = x + (size_t)bs * Dd + h * HD;
    xhb[wv][lane]      = xrow[lane];
    xhb[wv][lane + 32] = xrow[lane + 32];
  }
  wait_asynccnt0();
  __syncthreads();

  // phase B: s1 = q1 . c_keys[c]  (lane handles centroids lane+32*tt)
  float sc1[8];
#pragma unroll
  for (int tt = 0; tt < 8; ++tt) {
    const int c = lane + 32 * tt;
    const float* ck = &keybuf[c * SUBD];
    float a = 0.f;
#pragma unroll
    for (int d = 0; d < SUBD; ++d) a += qnb[wv][d] * ck[d];
    sc1[tt] = a;
  }
  __syncthreads();

  // phase C: async-restage with c_prime_keys, compute s2
#pragma unroll
  for (int i = t; i < SQRTN * SUBD / 4; i += 256)
    async_copy_b128(keybase + (unsigned)i * 16u,
                    (unsigned long long)(const void*)cpkeys + (unsigned long long)i * 16u);
  wait_asynccnt0();
  __syncthreads();

  float sc2[8];
#pragma unroll
  for (int tt = 0; tt < 8; ++tt) {
    const int c = lane + 32 * tt;
    const float* ck = &keybuf[c * SUBD];
    float a = 0.f;
#pragma unroll
    for (int d = 0; d < SUBD; ++d) a += qnb[wv][SUBD + d] * ck[d];
    sc2[tt] = a;
  }

  // top-16 per side. (Global top-16 of s1_i+s2_j lies inside top16 x top16,
  // which is a subset of the reference's top64 x top64 candidate set, so the
  // final top-16 is identical to the reference's.)
  float tv1[16], tv2[16]; int ti1[16], ti2[16];
  top16_256(sc1, lane, tv1, ti1);
  top16_256(sc2, lane, tv2, ti2);

  // joint 16x16 sums -> top-16
  float jsc[8];
#pragma unroll
  for (int tt = 0; tt < 8; ++tt) {
    const int m = lane + 32 * tt;
    jsc[tt] = tv1[m >> 4] + tv2[m & 15];
  }
  float fv[16]; int fi[16];
  top16_256(jsc, lane, fv, fi);

  int gid[KTOP];
#pragma unroll
  for (int r = 0; r < KTOP; ++r)
    gid[r] = ti1[fi[r] >> 4] * SQRTN + ti2[fi[r] & 15];

  // softmax over the 16 routing scores (redundant per lane, registers only)
  float mx = fv[0];
#pragma unroll
  for (int r = 1; r < KTOP; ++r) mx = fmaxf(mx, fv[r]);
  float rw[KTOP]; float se = 0.f;
#pragma unroll
  for (int r = 0; r < KTOP; ++r) { rw[r] = __expf(fv[r] - mx); se += rw[r]; }
  const float inv = 1.f / se;

  // expert FFN: hidden_k = gelu(x_head . wd_k);  out += rw_k*hidden_k * wu_k
  const float x0 = xhb[wv][2 * lane];
  const float x1 = xhb[wv][2 * lane + 1];
  float a0 = 0.f, a1 = 0.f;
#pragma unroll 1
  for (int r = 0; r < KTOP; ++r) {
    const size_t g = (size_t)gid[r] * HD;
    if (r + 1 < KTOP) {
      __builtin_prefetch(w_down + (size_t)gid[r + 1] * HD, 0, 1);
      __builtin_prefetch(w_up   + (size_t)gid[r + 1] * HD, 0, 1);
    }
    const float2 wd = ((const float2*)(w_down + g))[lane];
    float hp = wd.x * x0 + wd.y * x1;
#pragma unroll
    for (int o = 16; o >= 1; o >>= 1) hp += __shfl_xor(hp, o, 32);
    const float ge   = 0.5f * hp * (1.f + erff(hp * 0.70710678118654752f));
    const float coef = ge * rw[r] * inv;
    const float2 wu = ((const float2*)(w_up + g))[lane];
    a0 += coef * wu.x;
    a1 += coef * wu.y;
  }
  float2* yo = (float2*)(y + (size_t)bs * Dd + h * HD);
  yo[lane] = make_float2(a0, a1);
}

// ---------------------------------------------------------------------------
extern "C" void kernel_launch(void* const* d_in, const int* in_sizes, int n_in,
                              void* d_out, int out_size, void* d_ws, size_t ws_size,
                              hipStream_t stream) {
  const float* x    = (const float*)d_in[0];
  const float* Wq   = (const float*)d_in[1];
  const float* ln_g = (const float*)d_in[2];
  const float* ln_b = (const float*)d_in[3];
  const float* ck   = (const float*)d_in[4];
  const float* cpk  = (const float*)d_in[5];
  const float* wd   = (const float*)d_in[6];
  const float* wu   = (const float*)d_in[7];
  const float* Wout = (const float*)d_in[8];
  float* out = (float*)d_out;

  float* q = (float*)d_ws;                       // 4096*1024 f32 = 16 MB
  float* y = q + (size_t)MROWS * Dd;             // 4096*1024 f32 = 16 MB

  dim3 gGemm(Dd / 128, MROWS / 128);             // (8, 32)
  peer_gemm_bf16x3<<<gGemm, 256, 0, stream>>>(x, Wq, q, MROWS, Dd, Dd);
  peer_router_ffn<<<NTH / 8, 256, 0, stream>>>(x, q, ln_g, ln_b, ck, cpk, wd, wu, y);
  peer_gemm_bf16x3<<<gGemm, 256, 0, stream>>>(y, Wout, out, MROWS, Dd, Dd);
}
